// DecomposeWHAR_SingleSensor_v2_68341519614617
// MI455X (gfx1250) — matrix operand order repaired and checked by reading, not compiled, not run on hardware
//
#include <hip/hip_runtime.h>

// ---------------------------------------------------------------------------
// Types for CDNA5 WMMA (wave32): v_wmma_f32_16x16x32_bf16
// ---------------------------------------------------------------------------
typedef __bf16 bf16_t;
typedef __attribute__((ext_vector_type(16))) __bf16 v16bf;
typedef __attribute__((ext_vector_type(8)))  __bf16 v8bf;
typedef __attribute__((ext_vector_type(8)))  float  v8f;

// Problem constants (from reference)
#define BB   64           // batch
#define MM   12           // sensors folded into batch
#define LL   2048
#define DD   128          // model dim
#define TT   256          // L/S
#define DI   256          // 2*D mamba inner
#define DSN  16           // state dim
#define RKN  8            // dt rank
#define NTOK (BB*TT)      // 16384 mamba tokens

// ---------------------------------------------------------------------------
// WMMA fragment loaders (A/B use contiguous k order chosen here, not ISA table order):
//  A 16x32 (MxK): lanes 0-15 = rows, elems j -> K = kc + 16*hi + j (contiguous)
//  B 32x16 (KxN): lane n=l&15 col, elems j -> K = kc + 16*hi + j (contiguous)
//  C/D 16x16 f32: vgpr r, lane l: M = r + 8*(l>>4), N = l&15
// ---------------------------------------------------------------------------
__device__ __forceinline__ v16bf load_fragA(const bf16_t* base, int ld, int lane) {
  int row = lane & 15, hi = lane >> 4;
  const bf16_t* p = base + (size_t)row * ld + hi * 16;
  v8bf lo = *(const v8bf*)(p);
  v8bf hh = *(const v8bf*)(p + 8);
  v16bf r;
#pragma unroll
  for (int j = 0; j < 8; ++j) { r[j] = lo[j]; r[j + 8] = hh[j]; }
  return r;
}

__device__ __forceinline__ v16bf load_fragB(const bf16_t* base, int ld, int lane) {
  // base = &W[n0][k0], W row-major [N][K] so B[k][n]=W[n][k] is contiguous in k
  int n = lane & 15, hi = lane >> 4;
  return *(const v16bf*)(base + (size_t)n * ld + hi * 16);
}

#define WMMA_BF16(a, b, c) \
  __builtin_amdgcn_wmma_f32_16x16x32_bf16(false, (a), false, (b), (short)0, (c), false, false)

// ---------------------------------------------------------------------------
// fp32 -> bf16 weight conversion
// ---------------------------------------------------------------------------
__global__ void k_cvt(const float* __restrict__ s, bf16_t* __restrict__ d, int n) {
  int i = blockIdx.x * 256 + threadIdx.x;
  if (i < n) d[i] = (bf16_t)s[i];
}

__global__ void k_cvt_pad(const float* __restrict__ s, bf16_t* __restrict__ d,
                          int rows, int padrows, int cols) {
  int i = blockIdx.x * 256 + threadIdx.x;
  if (i < padrows * cols) {
    int r = i / cols;
    d[i] = (r < rows) ? (bf16_t)s[i] : (bf16_t)0.f;
  }
}

// ---------------------------------------------------------------------------
// Embedding: strided conv P=16, S=8, pad (4,4). u token-major: u[bb][t][d]
// ---------------------------------------------------------------------------
__global__ void k_embed(const float* __restrict__ xin, const float* __restrict__ ew,
                        const float* __restrict__ eb, float* __restrict__ u) {
  int bb = blockIdx.x, t = blockIdx.y, d = threadIdx.x;
  int b = bb / MM, m = bb % MM;
  __shared__ float win[16];
  if (d < 16) {
    int l = t * 8 - 4 + d;
    win[d] = (l >= 0 && l < LL) ? xin[(size_t)b * LL * MM + (size_t)l * MM + m] : 0.f;
  }
  __syncthreads();
  float acc = eb[d];
#pragma unroll
  for (int p = 0; p < 16; ++p) acc += ew[d * 16 + p] * win[p];
  u[((size_t)bb * TT + t) * DD + d] = acc;
}

// ---------------------------------------------------------------------------
// Depthwise conv K=5 + exact GELU -> bf16 v (token-major)
// ---------------------------------------------------------------------------
__global__ void k_dwgelu(const float* __restrict__ u, const float* __restrict__ dw,
                         const float* __restrict__ db, bf16_t* __restrict__ vh) {
  int bb = blockIdx.x, t = blockIdx.y, d = threadIdx.x;
  const float* ub = u + (size_t)bb * TT * DD;
  float acc = db[d];
#pragma unroll
  for (int j = 0; j < 5; ++j) {
    int tt = t - 2 + j;
    if (tt >= 0 && tt < TT) acc += dw[d * 5 + j] * ub[(size_t)tt * DD + d];
  }
  float g = 0.5f * acc * (1.f + erff(acc * 0.70710678118f));
  vh[((size_t)bb * TT + t) * DD + d] = (bf16_t)g;
}

// ---------------------------------------------------------------------------
// Pointwise conv as register-blocked WMMA GEMM:
//   w[bb][t][e] = sum_d pw[e][d] * v[bb][t][d] + pwb[e]
// One wave -> 16(t) x 64(e) tile: 1 A fragment reused against 4 B fragments,
// 4 f32 accumulators. K = 128 in chunks of 32. grid (768, 16), block 64.
// ---------------------------------------------------------------------------
__global__ void k_pw_gemm(const bf16_t* __restrict__ vh, const bf16_t* __restrict__ wt,
                          const float* __restrict__ pwb, float* __restrict__ wout) {
  int bb = blockIdx.x;
  int t0 = blockIdx.y * 16;
  int wv = threadIdx.x >> 5, lane = threadIdx.x & 31;
  int e0 = wv * 64;
  const bf16_t* A = vh + (size_t)bb * TT * DD + (size_t)t0 * DD;
  __builtin_prefetch(A + (size_t)(lane & 15) * DD, 0, 1);   // global_prefetch_b8
  v8f c0 = {}, c1 = {}, c2 = {}, c3 = {};
#pragma unroll
  for (int kc = 0; kc < DD; kc += 32) {
    v16bf a  = load_fragA(A + kc, DD, lane);
    v16bf b0 = load_fragB(wt + (size_t)(e0 +  0) * DD + kc, DD, lane);
    v16bf b1 = load_fragB(wt + (size_t)(e0 + 16) * DD + kc, DD, lane);
    v16bf b2 = load_fragB(wt + (size_t)(e0 + 32) * DD + kc, DD, lane);
    v16bf b3 = load_fragB(wt + (size_t)(e0 + 48) * DD + kc, DD, lane);
    c0 = WMMA_BF16(a, b0, c0);
    c1 = WMMA_BF16(a, b1, c1);
    c2 = WMMA_BF16(a, b2, c2);
    c3 = WMMA_BF16(a, b3, c3);
  }
  int n = lane & 15, hi = lane >> 4;
  float* W = wout + (size_t)bb * TT * DD;
#pragma unroll
  for (int s = 0; s < 4; ++s) {
    v8f c = (s == 0) ? c0 : (s == 1) ? c1 : (s == 2) ? c2 : c3;
    int e = e0 + 16 * s + n;
    float bias = pwb[e];
#pragma unroll
    for (int r = 0; r < 8; ++r)
      W[(size_t)(t0 + r + 8 * hi) * DD + e] = c[r] + bias;
  }
}

// ---------------------------------------------------------------------------
// Squeeze-excitation: s = mean_t w; a = sigmoid(w2 @ relu(w1 @ s + b1) + b2)
// grid 768, block 128
// ---------------------------------------------------------------------------
__global__ void k_se(const float* __restrict__ w, const float* __restrict__ w1,
                     const float* __restrict__ b1, const float* __restrict__ w2,
                     const float* __restrict__ b2, float* __restrict__ a_se) {
  int bb = blockIdx.x, e = threadIdx.x;
  const float* W = w + (size_t)bb * TT * DD;
  float s = 0.f;
  for (int t = 0; t < TT; ++t) s += W[(size_t)t * DD + e];
  s *= (1.f / (float)TT);
  __shared__ float sb[DD];
  __shared__ float hb[32];
  sb[e] = s;
  __syncthreads();
  if (e < 32) {
    float h = b1[e];
    for (int d = 0; d < DD; ++d) h += w1[e * DD + d] * sb[d];
    hb[e] = fmaxf(h, 0.f);
  }
  __syncthreads();
  float aa = b2[e];
#pragma unroll
  for (int r = 0; r < 32; ++r) aa += w2[e * 32 + r] * hb[r];
  a_se[bb * DD + e] = 1.f / (1.f + expf(-aa));
}

__global__ void k_resid(float* __restrict__ u, const float* __restrict__ w,
                        const float* __restrict__ a_se) {
  int bb = blockIdx.x, t = blockIdx.y, d = threadIdx.x;
  size_t idx = ((size_t)bb * TT + t) * DD + d;
  u[idx] += w[idx] * a_se[bb * DD + d];
}

// ---------------------------------------------------------------------------
// Mean over sensors -> xm[b][t][d] (f32 + bf16 copy). grid (64,256) block 128
// ---------------------------------------------------------------------------
__global__ void k_mean(const float* __restrict__ u, float* __restrict__ xm,
                       bf16_t* __restrict__ xmh) {
  int b = blockIdx.x, t = blockIdx.y, d = threadIdx.x;
  float s = 0.f;
#pragma unroll
  for (int m = 0; m < MM; ++m)
    s += u[(((size_t)(b * MM + m)) * TT + t) * DD + d];
  s *= (1.f / (float)MM);
  size_t idx = ((size_t)b * TT + t) * DD + d;
  xm[idx] = s;
  xmh[idx] = (bf16_t)s;
}

// ---------------------------------------------------------------------------
// Generic token GEMM, register-blocked: C[tok][n] = sum_k A[tok][k] * Wt[n][k]
// A: NTOK x K bf16 row-major; Wt: Npad x K bf16 row-major (rows >= Nout zeroed)
// One wave -> 16(tok) x up-to-64(n); block 128 (4 waves) covers 256 cols.
// grid (NTOK/16, ceil(Npad/256))
// ---------------------------------------------------------------------------
__global__ void k_tok_gemm(const bf16_t* __restrict__ A, const bf16_t* __restrict__ Wt,
                           float* __restrict__ C, int K, int Npad, int Nout, int ldc) {
  int tok0 = blockIdx.x * 16;
  int wv = threadIdx.x >> 5, lane = threadIdx.x & 31;
  int n0 = (blockIdx.y * 4 + wv) * 64;
  if (n0 >= Npad) return;                       // wave-uniform
  int nt = (Npad - n0) >> 4; if (nt > 4) nt = 4; // active 16-col sub-tiles
  __builtin_prefetch(A + (size_t)(tok0 + (lane & 15)) * K, 0, 1);
  v8f c0 = {}, c1 = {}, c2 = {}, c3 = {};
  for (int kc = 0; kc < K; kc += 32) {
    v16bf a = load_fragA(A + (size_t)tok0 * K + kc, K, lane);
    {            v16bf b = load_fragB(Wt + (size_t)(n0 +  0) * K + kc, K, lane);
                 c0 = WMMA_BF16(a, b, c0); }
    if (nt > 1){ v16bf b = load_fragB(Wt + (size_t)(n0 + 16) * K + kc, K, lane);
                 c1 = WMMA_BF16(a, b, c1); }
    if (nt > 2){ v16bf b = load_fragB(Wt + (size_t)(n0 + 32) * K + kc, K, lane);
                 c2 = WMMA_BF16(a, b, c2); }
    if (nt > 3){ v16bf b = load_fragB(Wt + (size_t)(n0 + 48) * K + kc, K, lane);
                 c3 = WMMA_BF16(a, b, c3); }
  }
  int n = lane & 15, hi = lane >> 4;
#pragma unroll
  for (int s = 0; s < 4; ++s) {
    if (s >= nt) break;
    v8f c = (s == 0) ? c0 : (s == 1) ? c1 : (s == 2) ? c2 : c3;
    int col = n0 + 16 * s + n;
    if (col < Nout) {
#pragma unroll
      for (int r = 0; r < 8; ++r)
        C[(size_t)(tok0 + r + 8 * hi) * ldc + col] = c[r];
    }
  }
}

// ---------------------------------------------------------------------------
// Mamba causal conv (k=4, pad 3 left) + SiLU. xz[b][t][512]; xi = first 256.
// grid (64,256), block 256
// ---------------------------------------------------------------------------
__global__ void k_mconv(const float* __restrict__ xz, const float* __restrict__ cw,
                        const float* __restrict__ cb, float* __restrict__ xcf,
                        bf16_t* __restrict__ xch) {
  int b = blockIdx.x, t = blockIdx.y, e = threadIdx.x;
  float acc = cb[e];
#pragma unroll
  for (int j = 0; j < 4; ++j) {
    int tt = t - 3 + j;
    if (tt >= 0) acc += cw[e * 4 + j] * xz[((size_t)b * TT + tt) * 512 + e];
  }
  float s = acc / (1.f + expf(-acc));
  size_t idx = ((size_t)b * TT + t) * DI + e;
  xcf[idx] = s;
  xch[idx] = (bf16_t)s;
}

// ---------------------------------------------------------------------------
// Selective scan. One thread per (b, d). grid 64, block 256.
// dbl[b][t][40] = {dtr[8], B[16], C[16]} broadcast via LDS each step.
// ---------------------------------------------------------------------------
__global__ void k_scan(const float* __restrict__ dbl, const float* __restrict__ xcf,
                       const float* __restrict__ xz, const float* __restrict__ dtw,
                       const float* __restrict__ dtb, const float* __restrict__ Alog,
                       const float* __restrict__ Dp, bf16_t* __restrict__ yh) {
  int b = blockIdx.x, d = threadIdx.x;
  float Ad[DSN];
#pragma unroll
  for (int s = 0; s < DSN; ++s) Ad[s] = -expf(Alog[d * DSN + s]);
  float wdt[RKN];
#pragma unroll
  for (int r = 0; r < RKN; ++r) wdt[r] = dtw[d * RKN + r];
  float bias = dtb[d], Dd = Dp[d];
  float h[DSN];
#pragma unroll
  for (int s = 0; s < DSN; ++s) h[s] = 0.f;
  __shared__ float sh[40];
  for (int t = 0; t < TT; ++t) {
    __syncthreads();
    if (d < 40) sh[d] = dbl[((size_t)b * TT + t) * 40 + d];
    __syncthreads();
    float dtv = bias;
#pragma unroll
    for (int r = 0; r < RKN; ++r) dtv += wdt[r] * sh[r];
    dtv = (dtv > 20.f) ? dtv : log1pf(expf(dtv));   // softplus
    float xc = xcf[((size_t)b * TT + t) * DI + d];
    float acc = 0.f;
#pragma unroll
    for (int s = 0; s < DSN; ++s) {
      h[s] = expf(dtv * Ad[s]) * h[s] + dtv * sh[8 + s] * xc;
      acc += h[s] * sh[24 + s];
    }
    float y = acc + Dd * xc;
    float zv = xz[((size_t)b * TT + t) * 512 + DI + d];
    y *= zv / (1.f + expf(-zv));                    // * silu(z)
    yh[((size_t)b * TT + t) * DI + d] = (bf16_t)y;
  }
}

// ---------------------------------------------------------------------------
// Residual + LayerNorm over d=128. grid NTOK, block 128. Writes xm + bf16 copy.
// ---------------------------------------------------------------------------
__global__ void k_lnres(float* __restrict__ xm, const float* __restrict__ mo,
                        const float* __restrict__ g, const float* __restrict__ be,
                        bf16_t* __restrict__ xmh) {
  int tok = blockIdx.x, d = threadIdx.x;
  size_t idx = (size_t)tok * DD + d;
  float val = xm[idx] + mo[idx];
  __shared__ float red[DD];
  red[d] = val;
  __syncthreads();
  for (int s = 64; s > 0; s >>= 1) { if (d < s) red[d] += red[d + s]; __syncthreads(); }
  float mu = red[0] * (1.f / (float)DD);
  __syncthreads();
  float dv = val - mu;
  red[d] = dv * dv;
  __syncthreads();
  for (int s = 64; s > 0; s >>= 1) { if (d < s) red[d] += red[d + s]; __syncthreads(); }
  float var = red[0] * (1.f / (float)DD);
  float o = dv * rsqrtf(var + 1e-5f) * g[d] + be[d];
  xm[idx] = o;
  xmh[idx] = (bf16_t)o;
}

// ---------------------------------------------------------------------------
// xflat writeout: out[b*32768 + d*256 + t] = xm[b][t][d]. grid (64,128) block 256
// ---------------------------------------------------------------------------
__global__ void k_flat(const float* __restrict__ xm, float* __restrict__ out) {
  int b = blockIdx.x, d = blockIdx.y, t = threadIdx.x;
  out[(size_t)b * (DD * TT) + (size_t)d * TT + t] = xm[((size_t)b * TT + t) * DD + d];
}

// ---------------------------------------------------------------------------
// Final FC: pred[b][c] = xflat[b] . fc_w[c] + fc_b[c]. grid (64,18), block 256
// ---------------------------------------------------------------------------
__global__ void k_fc(const float* __restrict__ out, const float* __restrict__ fcw,
                     const float* __restrict__ fcb, float* __restrict__ pred) {
  int b = blockIdx.x, c = blockIdx.y, tid = threadIdx.x;
  const float* xf = out + (size_t)b * (DD * TT);
  const float* wr = fcw + (size_t)c * (DD * TT);
  float acc = 0.f;
  for (int i = tid; i < DD * TT; i += 256) acc += xf[i] * wr[i];
  __shared__ float red[256];
  red[tid] = acc;
  __syncthreads();
  for (int s = 128; s > 0; s >>= 1) { if (tid < s) red[tid] += red[tid + s]; __syncthreads(); }
  if (tid == 0) pred[(size_t)b * 18 + c] = red[0] + fcb[c];
}

// ---------------------------------------------------------------------------
// Workspace layout (bytes). Region A (conv u) is reused for mamba buffers,
// region B (conv w) is reused for xm/xmh.
// ---------------------------------------------------------------------------
static const size_t SZ_U    = (size_t)768 * TT * DD * 4;   // 100,663,296
static const size_t OFF_A   = 0;
static const size_t OFF_B   = SZ_U;
static const size_t OFF_VH  = OFF_B + SZ_U;
static const size_t SZ_VH   = (size_t)768 * TT * DD * 2;
static const size_t OFF_SM  = OFF_VH + SZ_VH;
// region A reuse (mamba):
static const size_t OFF_XZ  = OFF_A;                                   // 16384*512*4
static const size_t OFF_XCF = OFF_XZ  + (size_t)NTOK * 512 * 4;        // 16384*256*4
static const size_t OFF_XCH = OFF_XCF + (size_t)NTOK * DI * 4;         // 16384*256*2
static const size_t OFF_DBL = OFF_XCH + (size_t)NTOK * DI * 2;         // 16384*40*4
static const size_t OFF_YH  = OFF_DBL + (size_t)NTOK * 40 * 4;         // 16384*256*2
static const size_t OFF_MO  = OFF_YH  + (size_t)NTOK * DI * 2;         // 16384*128*4
// region B reuse:
static const size_t OFF_XM  = OFF_B;                                   // 16384*128*4
static const size_t OFF_XMH = OFF_XM + (size_t)NTOK * DD * 4;          // 16384*128*2
// small weight staging:
static const size_t OFF_PWH   = OFF_SM;                 // 128*128*2
static const size_t OFF_ASE   = OFF_PWH   + 32768;      // 768*128*4
static const size_t OFF_INWH  = OFF_ASE   + 393216;     // 512*128*2
static const size_t OFF_XPJH  = OFF_INWH  + 131072;     // 48*256*2
static const size_t OFF_OUTWH = OFF_XPJH  + 24576;      // 128*256*2

extern "C" void kernel_launch(void* const* d_in, const int* in_sizes, int n_in,
                              void* d_out, int out_size, void* d_ws, size_t ws_size,
                              hipStream_t stream) {
  (void)in_sizes; (void)n_in; (void)out_size; (void)ws_size;
  const float* inputs   = (const float*)d_in[0];
  const float* emb_w    = (const float*)d_in[1];
  const float* emb_b    = (const float*)d_in[2];
  const float* dw_w     = (const float*)d_in[3];
  const float* dw_b     = (const float*)d_in[4];
  const float* pw_w     = (const float*)d_in[5];
  const float* pw_b     = (const float*)d_in[6];
  const float* se_w1    = (const float*)d_in[7];
  const float* se_b1    = (const float*)d_in[8];
  const float* se_w2    = (const float*)d_in[9];
  const float* se_b2    = (const float*)d_in[10];
  const float* in_w     = (const float*)d_in[11];
  const float* conv_w   = (const float*)d_in[12];
  const float* conv_b   = (const float*)d_in[13];
  const float* xproj_w  = (const float*)d_in[14];
  const float* dt_w     = (const float*)d_in[15];
  const float* dt_b     = (const float*)d_in[16];
  const float* Alog     = (const float*)d_in[17];
  const float* Dp       = (const float*)d_in[18];
  const float* out_w    = (const float*)d_in[19];
  const float* ln_g     = (const float*)d_in[20];
  const float* ln_b     = (const float*)d_in[21];
  const float* fc_w     = (const float*)d_in[22];
  const float* fc_b     = (const float*)d_in[23];

  uint8_t* ws = (uint8_t*)d_ws;
  float*  u     = (float*)(ws + OFF_A);
  float*  wbuf  = (float*)(ws + OFF_B);
  bf16_t* vh    = (bf16_t*)(ws + OFF_VH);
  bf16_t* pw_wh = (bf16_t*)(ws + OFF_PWH);
  float*  a_se  = (float*)(ws + OFF_ASE);
  float*  xm    = (float*)(ws + OFF_XM);
  bf16_t* xmh   = (bf16_t*)(ws + OFF_XMH);
  bf16_t* in_wh = (bf16_t*)(ws + OFF_INWH);
  float*  xz    = (float*)(ws + OFF_XZ);
  float*  xcf   = (float*)(ws + OFF_XCF);
  bf16_t* xch   = (bf16_t*)(ws + OFF_XCH);
  bf16_t* xpj_h = (bf16_t*)(ws + OFF_XPJH);
  float*  dbl   = (float*)(ws + OFF_DBL);
  bf16_t* yh    = (bf16_t*)(ws + OFF_YH);
  bf16_t* out_wh= (bf16_t*)(ws + OFF_OUTWH);
  float*  mo    = (float*)(ws + OFF_MO);
  float*  outp  = (float*)d_out;

  // 1) embedding
  k_embed<<<dim3(768, TT), 128, 0, stream>>>(inputs, emb_w, emb_b, u);

  // 2) three conv-SE blocks (pointwise GEMM on WMMA)
  for (int i = 0; i < 3; ++i) {
    k_cvt<<<64, 256, 0, stream>>>(pw_w + (size_t)i * 16384, pw_wh, 16384);
    k_dwgelu<<<dim3(768, TT), 128, 0, stream>>>(u, dw_w + (size_t)i * 640,
                                                dw_b + (size_t)i * 128, vh);
    k_pw_gemm<<<dim3(768, 16), 64, 0, stream>>>(vh, pw_wh,
                                                pw_b + (size_t)i * 128, wbuf);
    k_se<<<768, 128, 0, stream>>>(wbuf, se_w1 + (size_t)i * 4096, se_b1 + (size_t)i * 32,
                                  se_w2 + (size_t)i * 4096, se_b2 + (size_t)i * 128, a_se);
    k_resid<<<dim3(768, TT), 128, 0, stream>>>(u, wbuf, a_se);
  }

  // 3) sensor mean -> token stream xm (f32 + bf16)
  k_mean<<<dim3(BB, TT), 128, 0, stream>>>(u, xm, xmh);

  // 4) two mamba blocks
  for (int i = 0; i < 2; ++i) {
    k_cvt<<<256, 256, 0, stream>>>(in_w + (size_t)i * 65536, in_wh, 65536);
    // xz = tokens(16384x128) @ in_w^T(128x512)
    k_tok_gemm<<<dim3(NTOK / 16, 2), 128, 0, stream>>>(xmh, in_wh, xz, DD, 512, 512, 512);
    k_mconv<<<dim3(BB, TT), DI, 0, stream>>>(xz, conv_w + (size_t)i * 1024,
                                             conv_b + (size_t)i * 256, xcf, xch);
    k_cvt_pad<<<(48 * 256 + 255) / 256, 256, 0, stream>>>(xproj_w + (size_t)i * 10240,
                                                          xpj_h, 40, 48, 256);
    // dbl = xc(16384x256) @ xproj^T(256x40)
    k_tok_gemm<<<dim3(NTOK / 16, 1), 128, 0, stream>>>(xch, xpj_h, dbl, DI, 48, 40, 40);
    k_scan<<<BB, DI, 0, stream>>>(dbl, xcf, xz, dt_w + (size_t)i * 2048,
                                  dt_b + (size_t)i * 256, Alog + (size_t)i * 4096,
                                  Dp + (size_t)i * 256, yh);
    k_cvt<<<128, 256, 0, stream>>>(out_w + (size_t)i * 32768, out_wh, 32768);
    // mo = y(16384x256) @ out_w^T(256x128)
    k_tok_gemm<<<dim3(NTOK / 16, 1), 128, 0, stream>>>(yh, out_wh, mo, DI, DD, DD, DD);
    k_lnres<<<NTOK, DD, 0, stream>>>(xm, mo, ln_g + (size_t)i * 128,
                                     ln_b + (size_t)i * 128, xmh);
  }

  // 5) xflat writeout + FC head
  k_flat<<<dim3(BB, DD), TT, 0, stream>>>(xm, outp);
  k_fc<<<dim3(BB, 18), 256, 0, stream>>>(outp, fc_w, fc_b, outp + (size_t)BB * DD * TT);
}
